// CgpHmmLayer_6545530159761
// MI455X (gfx1250) — compile-verified, matrix operand order checked by reading, not run on hardware
//
#include <hip/hip_runtime.h>

// ---------------------------------------------------------------------------
// HMM forward (log-likelihood) on MI455X / gfx1250.
//   BATCH=64, T=4096, ALPHABET=32, N_STATES=128.
// Sequential scan over T; per step: alpha <- (alpha @ A) * E_t, with deferred
// renormalization every 8 steps. Matmul via v_wmma_f32_16x16x32_bf16:
// 4 workgroups (16 batch rows each) x 8 waves (one 16-state N-tile each).
// Transition-matrix WMMA B-operands live in VGPRs for the whole T loop.
// Emission gather for step t+1 is software-pipelined under step t's WMMAs.
// ---------------------------------------------------------------------------

#define BATCH   64
#define T_LEN   4096
#define ALPHA_N 32
#define S_N     128
#define TILE_T  64   // token staging tile (steps)
#define NS      8    // renormalize every NS steps (deferred scaling)

typedef __attribute__((ext_vector_type(16))) __bf16 v16bf;
typedef __attribute__((ext_vector_type(8)))  float  v8f;
typedef __attribute__((ext_vector_type(4)))  float  f4;

union FragBF { v16bf v; f4 q[2]; };   // 32 bytes: one bf16 WMMA A/B fragment

__device__ __forceinline__ unsigned short f2bf_rne(float f) {
  // round-to-nearest-even f32 -> bf16 (used only in one-time preamble)
  union { float f; unsigned u; } c; c.f = f;
  unsigned r = c.u + 0x7FFFu + ((c.u >> 16) & 1u);
  return (unsigned short)(r >> 16);
}

__device__ __forceinline__ unsigned short f2bf_trunc(float f) {
  // truncating f32 -> bf16: just the high 16 bits (cheap: shift or d16_hi store)
  union { float f; unsigned u; } c; c.f = f;
  return (unsigned short)(c.u >> 16);
}

// ---------------------------------------------------------------------------
// Pass 1: collapse one-hot observations [B,T,A] -> token ids, time-major [T,B]
// ---------------------------------------------------------------------------
__global__ void hmm_tokens_kernel(const float* __restrict__ inputs,
                                  int* __restrict__ tokens) {
  int idx = blockIdx.x * blockDim.x + threadIdx.x;   // b*T + t
  if (idx >= BATCH * T_LEN) return;
  int b = idx / T_LEN, t = idx % T_LEN;
  const float* p = inputs + (size_t)idx * ALPHA_N;
  int tok = 0; float best = p[0];
  #pragma unroll
  for (int i = 1; i < ALPHA_N; ++i) { float v = p[i]; if (v > best) { best = v; tok = i; } }
  tokens[t * BATCH + b] = tok;
}

// ---------------------------------------------------------------------------
// Pass 2: the forward scan. grid = 4 workgroups, 256 threads (8 waves) each.
// ---------------------------------------------------------------------------
__global__ __launch_bounds__(256) void hmm_forward_kernel(
    const int*   __restrict__ tokens,
    const float* __restrict__ A_logits,
    const float* __restrict__ B_logits,
    const float* __restrict__ I_logits,
    float*       __restrict__ out)
{
  __shared__ __align__(16) unsigned short sAT[S_N * S_N];     // A^T bf16: [to][from] (32KB)
  __shared__ __align__(16) float          sBm[ALPHA_N * S_N]; // emissions f32 (16KB)
  __shared__ __align__(16) unsigned short sA[2][16 * S_N];    // alpha bf16, double buffered (8KB)
  __shared__ __align__(16) int            sTok[TILE_T * 16];  // staged tokens (4KB)
  __shared__ __align__(16) float          sPart[16][8];       // per-wave row sums
  __shared__               float          sI[S_N];            // initial distribution

  const int tid  = threadIdx.x;
  const int lane = tid & 31;        // wave32
  const int w    = tid >> 5;        // wave id 0..7 -> N-tile
  const int hf   = lane >> 4;       // half-wave
  const int ln   = lane & 15;
  const int n0   = w * 16;          // this wave's state tile
  const int b0   = blockIdx.x * 16; // this block's batch rows

  // --- one-time softmaxes into LDS -----------------------------------------
  if (tid < S_N) {
    // rows of A softmaxed (axis=1), written transposed as bf16
    const f4* row4 = (const f4*)(A_logits + tid * S_N);
    float mx = -3.4e38f;
    for (int j = 0; j < S_N / 4; ++j) {
      f4 v = row4[j];
      mx = fmaxf(mx, fmaxf(fmaxf(v.x, v.y), fmaxf(v.z, v.w)));
    }
    float sum = 0.f;
    for (int j = 0; j < S_N / 4; ++j) {
      f4 v = row4[j];
      sum += __expf(v.x - mx) + __expf(v.y - mx) + __expf(v.z - mx) + __expf(v.w - mx);
    }
    float inv = 1.f / sum;
    for (int j = 0; j < S_N / 4; ++j) {
      f4 v = row4[j];
      sAT[(j * 4 + 0) * S_N + tid] = f2bf_rne(__expf(v.x - mx) * inv);
      sAT[(j * 4 + 1) * S_N + tid] = f2bf_rne(__expf(v.y - mx) * inv);
      sAT[(j * 4 + 2) * S_N + tid] = f2bf_rne(__expf(v.z - mx) * inv);
      sAT[(j * 4 + 3) * S_N + tid] = f2bf_rne(__expf(v.w - mx) * inv);
    }
    // columns of B softmaxed over the alphabet (axis=0)
    float bmx = B_logits[tid];
    for (int a = 1; a < ALPHA_N; ++a) bmx = fmaxf(bmx, B_logits[a * S_N + tid]);
    float bsum = 0.f;
    for (int a = 0; a < ALPHA_N; ++a) bsum += __expf(B_logits[a * S_N + tid] - bmx);
    float binv = 1.f / bsum;
    for (int a = 0; a < ALPHA_N; ++a)
      sBm[a * S_N + tid] = __expf(B_logits[a * S_N + tid] - bmx) * binv;
    // initial distribution (each thread redundantly reduces 128 values)
    float imx = I_logits[0];
    for (int j = 1; j < S_N; ++j) imx = fmaxf(imx, I_logits[j]);
    float isum = 0.f;
    for (int j = 0; j < S_N; ++j) isum += __expf(I_logits[j] - imx);
    sI[tid] = __expf(I_logits[tid] - imx) / isum;
  }
  __syncthreads();

  // --- loop-invariant WMMA B-operands: A^T tile K(128) x N(16), bf16 -------
  // lane ln holds column n0+ln; elems 0..7 = K{kc*32 + hf*8 ..},
  // elems 8..15 = K{kc*32 + 16 + hf*8 ..}  (ISA 16-bit operand layout)
  FragBF bfrag[4];
  #pragma unroll
  for (int kc = 0; kc < 4; ++kc) {
    int e0 = (n0 + ln) * S_N + kc * 32 + hf * 8;
    bfrag[kc].q[0] = *(const f4*)&sAT[e0];
    bfrag[kc].q[1] = *(const f4*)&sAT[e0 + 16];
  }

  // --- alpha(t=0) = I * Bm[token] (unnormalized; deferred scaling covers it)
  {
    int m = tid >> 4, kg = tid & 15;
    int tok = tokens[b0 + m];          // time-major: step 0, row b0+m
    #pragma unroll
    for (int k = 0; k < 8; ++k) {
      int kk = kg * 8 + k;
      sA[0][m * S_N + kk] = f2bf_trunc(sI[kk] * sBm[tok * S_N + kk]);
    }
  }
  __syncthreads();

  float loglik = 0.f;   // owned by threads 0..15 (row = tid)
  int   cur = 0;
  int   since = 0;

  for (int t0 = 1; t0 < T_LEN; t0 += TILE_T) {
    int te  = min(t0 + TILE_T, T_LEN);
    int cnt = (te - t0) * 16;
    // stage token tile (prev step-end barrier protects against WAR)
    for (int i = tid; i < cnt; i += 256)
      sTok[i] = tokens[(t0 + (i >> 4)) * BATCH + b0 + (i & 15)];
    __syncthreads();

    // prime the emission-probability pipeline for step t0
    float bmv[8];
    #pragma unroll
    for (int r = 0; r < 8; ++r) {
      int tok = sTok[r + 8 * hf];
      bmv[r] = sBm[tok * S_N + n0 + ln];
    }

    for (int t = t0; t < te; ++t) {
      const unsigned short* al = sA[cur];
      unsigned short*       an = sA[cur ^ 1];

      // alpha-fragment loads + 4 chained WMMAs (K = 128)
      v8f acc = {0.f, 0.f, 0.f, 0.f, 0.f, 0.f, 0.f, 0.f};
      #pragma unroll
      for (int kc = 0; kc < 4; ++kc) {
        FragBF af;
        int e0 = ln * S_N + kc * 32 + hf * 8;   // row m=ln of alpha
        af.q[0] = *(const f4*)&al[e0];
        af.q[1] = *(const f4*)&al[e0 + 16];
        acc = __builtin_amdgcn_wmma_f32_16x16x32_bf16(
            false, af.v, false, bfrag[kc].v, (short)0, acc, false, false);
      }

      // prefetch next step's emission probs (independent of alpha; overlaps
      // the WMMA chain and fills WMMA->VALU hazard slots). Clamped at tile
      // end; re-primed after the next staging barrier anyway.
      int tn = (t + 1 < te) ? (t + 1 - t0) : 0;
      float bmn[8];
      #pragma unroll
      for (int r = 0; r < 8; ++r) {
        int tok = sTok[tn * 16 + r + 8 * hf];
        bmn[r] = sBm[tok * S_N + n0 + ln];
      }

      // emission multiply: acc[r] holds (m = r + 8*hf, n = n0 + ln)
      #pragma unroll
      for (int r = 0; r < 8; ++r) acc[r] *= bmv[r];

      ++since;
      bool donorm = (since == NS) || (t == T_LEN - 1);
      if (donorm) {
        // per-wave partial row sums via half-wave xor shuffles
        #pragma unroll
        for (int r = 0; r < 8; ++r) {
          float s = acc[r];
          s += __shfl_xor(s, 1);
          s += __shfl_xor(s, 2);
          s += __shfl_xor(s, 4);
          s += __shfl_xor(s, 8);
          if (ln == 0) sPart[r + 8 * hf][w] = s;
        }
        __syncthreads();
        // totals, rescale, write rescaled bf16 alpha
        #pragma unroll
        for (int r = 0; r < 8; ++r) {
          int m = r + 8 * hf;
          f4 p0 = *(const f4*)&sPart[m][0];
          f4 p1 = *(const f4*)&sPart[m][4];
          float tot = (p0.x + p0.y) + (p0.z + p0.w) + (p1.x + p1.y) + (p1.z + p1.w);
          acc[r] *= 1.f / tot;
          an[m * S_N + n0 + ln] = f2bf_trunc(acc[r]);
        }
        // log-likelihood of the deferred window, one owner thread per row
        if (tid < 16) {
          f4 p0 = *(const f4*)&sPart[tid][0];
          f4 p1 = *(const f4*)&sPart[tid][4];
          float tot = (p0.x + p0.y) + (p0.z + p0.w) + (p1.x + p1.y) + (p1.z + p1.w);
          loglik += __logf(tot);
        }
        since = 0;
      } else {
        #pragma unroll
        for (int r = 0; r < 8; ++r) {
          int m = r + 8 * hf;
          an[m * S_N + n0 + ln] = f2bf_trunc(acc[r]);
        }
      }

      #pragma unroll
      for (int r = 0; r < 8; ++r) bmv[r] = bmn[r];

      cur ^= 1;
      __syncthreads();
    }
  }

  if (tid < 16) out[b0 + tid] = loglik;
}

// ---------------------------------------------------------------------------
extern "C" void kernel_launch(void* const* d_in, const int* in_sizes, int n_in,
                              void* d_out, int out_size, void* d_ws, size_t ws_size,
                              hipStream_t stream) {
  const float* inputs   = (const float*)d_in[0];  // [64,4096,32] one-hot f32
  const float* A_logits = (const float*)d_in[1];  // [128,128]
  const float* B_logits = (const float*)d_in[2];  // [32,128]
  const float* I_logits = (const float*)d_in[3];  // [128]
  float* out    = (float*)d_out;                  // [64] f32 loglik
  int*   tokens = (int*)d_ws;                     // [T,BATCH] = 1 MB scratch

  hmm_tokens_kernel<<<(BATCH * T_LEN + 255) / 256, 256, 0, stream>>>(inputs, tokens);
  hmm_forward_kernel<<<BATCH / 16, 256, 0, stream>>>(tokens, A_logits, B_logits,
                                                     I_logits, out);
}
